// InternVL3VisualModel_12360915878696
// MI455X (gfx1250) — matrix operand order repaired and checked by reading, not compiled, not run on hardware
//
#include <hip/hip_runtime.h>
#include <hip/hip_bf16.h>

typedef __attribute__((ext_vector_type(16))) _Float16 v16h;
typedef __attribute__((ext_vector_type(8)))  _Float16 v8h;
typedef __attribute__((ext_vector_type(8)))  float    v8f;

// ---------------------------------------------------------------------------
// model dims
#define VD    1024          // hidden
#define VH    16            // heads
#define VHD   64            // head dim
#define VI    4096          // ffn
#define VB    4             // image tiles
#define VN    1025          // tokens incl CLS
#define VM    (VB*VN)       // 4100 token rows
#define KPAT  608           // 588 padded to 19*32
#define LLMD  2048

// ---------------------------------------------------------------------------
// WMMA helpers — CDNA5 V_WMMA_F32_16X16X32_F16
__device__ __forceinline__ v8f wmma16(v16h a, v16h b, v8f c) {
  return __builtin_amdgcn_wmma_f32_16x16x32_f16(false, a, false, b, (short)0, c,
                                                false, false);
}

// A fragment (16x32 f16, M x K). Lanes 0-15 hold M=lane, K=0..15;
// lanes 16-31 hold M=lane-16, K=16..31 (contiguous halves).
__device__ __forceinline__ v16h frag_a(const _Float16* base, int row0, int kb, int ld) {
  int lane = threadIdx.x & 31;
  const _Float16* p = base + (size_t)(row0 + (lane & 15)) * ld + kb + ((lane < 16) ? 0 : 16);
  v8h lo = *(const v8h*)p;
  v8h hi = *(const v8h*)(p + 8);
  v16h r;
#pragma unroll
  for (int i = 0; i < 8; ++i) { r[i] = lo[i]; r[8 + i] = hi[i]; }
  return r;
}

// B fragment (32x16 f16, K x N), source stored N-major [n][k] (weight layout
// W[n,k] so B[k][n] = W[n][k]). Lanes 0-15: N=lane, K=0..15; lanes 16-31:
// N=lane-16, K=16..31 (contiguous halves).
__device__ __forceinline__ v16h frag_b(const _Float16* base, int col0, int kb, int ld) {
  int lane = threadIdx.x & 31;
  const _Float16* p = base + (size_t)(col0 + (lane & 15)) * ld + kb + ((lane < 16) ? 0 : 16);
  v8h lo = *(const v8h*)p;
  v8h hi = *(const v8h*)(p + 8);
  v16h r;
#pragma unroll
  for (int i = 0; i < 8; ++i) { r[i] = lo[i]; r[8 + i] = hi[i]; }
  return r;
}

// B fragment from a K-major [k][n] LDS tile (used for attention V).
__device__ __forceinline__ v16h frag_b_kmajor(const _Float16* base, int col0, int ld) {
  int lane = threadIdx.x & 31;
  int n = col0 + (lane & 15);
  int koff = (lane < 16) ? 0 : 16;
  v16h r;
#pragma unroll
  for (int i = 0; i < 16; ++i) r[i] = base[(size_t)(koff + i) * ld + n];
  return r;
}

__device__ __forceinline__ float gelu_exact(float v) {
  return 0.5f * v * (1.0f + erff(v * 0.70710678118654752f));
}

// ---------------------------------------------------------------------------
// elementwise / data-movement kernels
__global__ void k_cvt(const float* __restrict__ s, _Float16* __restrict__ d, int n) {
  int i = (blockIdx.x * 256 + threadIdx.x) * 4;
#pragma unroll
  for (int t = 0; t < 4; ++t)
    if (i + t < n) d[i + t] = (_Float16)s[i + t];
}

__global__ void k_cvt_patchw(const float* __restrict__ s, _Float16* __restrict__ d) {
  int idx = blockIdx.x * 256 + threadIdx.x;          // 1024*608
  if (idx >= VD * KPAT) return;
  int n = idx / KPAT, k = idx % KPAT;
  d[idx] = (k < 588) ? (_Float16)s[n * 588 + k] : (_Float16)0.0f;
}

__global__ void k_im2col(const float* __restrict__ pix, _Float16* __restrict__ a) {
  int idx = blockIdx.x * 256 + threadIdx.x;          // 4096*608
  if (idx >= VB * 1024 * KPAT) return;
  int row = idx / KPAT, k = idx % KPAT;
  _Float16 v = (_Float16)0.0f;
  if (k < 588) {
    int cin = k / 196, rem = k % 196, py = rem / 14, px = rem % 14;
    int b = row >> 10, t = row & 1023, pr = t >> 5, pc = t & 31;
    v = (_Float16)pix[(((size_t)(b * 3 + cin)) * 448 + pr * 14 + py) * 448 + pc * 14 + px];
  }
  a[idx] = v;
}

__global__ void k_build_x(const float* __restrict__ cls, const float* __restrict__ pos,
                          const float* __restrict__ patchout, float* __restrict__ x) {
  int idx = blockIdx.x * 256 + threadIdx.x;          // 4*1025*1024
  if (idx >= VM * VD) return;
  int d = idx & 1023;
  int t = (idx >> 10) % VN;
  int b = idx / (VN * VD);
  float v = (t == 0) ? cls[d] : patchout[((size_t)(b * 1024 + (t - 1))) * VD + d];
  x[idx] = v + pos[t * VD + d];
}

// ---------------------------------------------------------------------------
// LayerNorm: fp32 row -> f16 row
__global__ __launch_bounds__(256) void k_layernorm(const float* __restrict__ x,
                                                   const float* __restrict__ w,
                                                   const float* __restrict__ b,
                                                   _Float16* __restrict__ out, int D) {
  __shared__ float sred[16];
  int row = blockIdx.x;
  const float* xr = x + (size_t)row * D;
  float s = 0.f, s2 = 0.f;
  for (int i = threadIdx.x; i < D; i += 256) { float v = xr[i]; s += v; s2 += v * v; }
#pragma unroll
  for (int off = 16; off; off >>= 1) {
    s += __shfl_xor(s, off, 32);
    s2 += __shfl_xor(s2, off, 32);
  }
  if ((threadIdx.x & 31) == 0) { sred[threadIdx.x >> 5] = s; sred[8 + (threadIdx.x >> 5)] = s2; }
  __syncthreads();
  if (threadIdx.x < 32) {
    float a = (threadIdx.x < 8) ? sred[threadIdx.x] : 0.f;
    float c = (threadIdx.x < 8) ? sred[8 + threadIdx.x] : 0.f;
#pragma unroll
    for (int off = 4; off; off >>= 1) { a += __shfl_xor(a, off, 32); c += __shfl_xor(c, off, 32); }
    if (threadIdx.x == 0) { sred[0] = a; sred[1] = c; }
  }
  __syncthreads();
  float mean = sred[0] / D;
  float var = sred[1] / D - mean * mean;
  float inv = rsqrtf(var + 1e-6f);
  for (int i = threadIdx.x; i < D; i += 256)
    out[(size_t)row * D + i] = (_Float16)((xr[i] - mean) * inv * w[i] + b[i]);
}

// Pixel-shuffle (drop CLS, 2x2 merge) + LayerNorm -> f16 [B*256, 4096]
__global__ __launch_bounds__(256) void k_shuffle_ln(const float* __restrict__ x,
                                                    const float* __restrict__ w,
                                                    const float* __restrict__ bb,
                                                    _Float16* __restrict__ out) {
  __shared__ float sred[16];
  const int CH = 4096;
  int row = blockIdx.x;                  // 0..1023
  int b = row >> 8, p = row & 255;
  int ii = p >> 4, jj = p & 15;
  float vals[16];
  float s = 0.f, s2 = 0.f;
#pragma unroll
  for (int t = 0; t < 16; ++t) {
    int c = threadIdx.x + t * 256;
    int sel = c >> 10, d = c & 1023;
    int s1 = sel >> 1, sq = sel & 1;
    int tok = 1 + (2 * ii + s1) * 32 + (2 * jj + sq);
    float v = x[((size_t)(b * VN + tok)) * VD + d];
    vals[t] = v; s += v; s2 += v * v;
  }
#pragma unroll
  for (int off = 16; off; off >>= 1) {
    s += __shfl_xor(s, off, 32);
    s2 += __shfl_xor(s2, off, 32);
  }
  if ((threadIdx.x & 31) == 0) { sred[threadIdx.x >> 5] = s; sred[8 + (threadIdx.x >> 5)] = s2; }
  __syncthreads();
  if (threadIdx.x < 32) {
    float a = (threadIdx.x < 8) ? sred[threadIdx.x] : 0.f;
    float c = (threadIdx.x < 8) ? sred[8 + threadIdx.x] : 0.f;
#pragma unroll
    for (int off = 4; off; off >>= 1) { a += __shfl_xor(a, off, 32); c += __shfl_xor(c, off, 32); }
    if (threadIdx.x == 0) { sred[0] = a; sred[1] = c; }
  }
  __syncthreads();
  float mean = sred[0] / CH;
  float var = sred[1] / CH - mean * mean;
  float inv = rsqrtf(var + 1e-6f);
#pragma unroll
  for (int t = 0; t < 16; ++t) {
    int c = threadIdx.x + t * 256;
    out[(size_t)row * CH + c] = (_Float16)((vals[t] - mean) * inv * w[c] + bb[c]);
  }
}

// ---------------------------------------------------------------------------
// Generic WMMA GEMM: C[M,N] = A[M,K] (f16 row-major) * W[N,K]^T (+epilogue)
// EP 0: outH = acc+bias (f16)
// EP 1: outH = gelu(acc+bias) (f16)
// EP 2: outF = resid + ls[n]*(acc+bias) (f32)
// EP 3: outF = acc+bias (f32)
#define TM 128
#define TN 128
#define LDT 40

template <int EP>
__global__ __launch_bounds__(256) void k_gemm(const _Float16* __restrict__ A,
                                              const _Float16* __restrict__ W,
                                              const float* __restrict__ bias,
                                              const float* __restrict__ ls,
                                              const float* __restrict__ resid,
                                              float* __restrict__ outF,
                                              _Float16* __restrict__ outH,
                                              int M, int N, int K) {
  __shared__ _Float16 sA[TM * LDT];
  __shared__ _Float16 sB[TN * LDT];
  int m0 = blockIdx.y * TM, n0 = blockIdx.x * TN;
  int wave = threadIdx.x >> 5;
  int wm = wave >> 1, wn = wave & 1;           // 4 x 2 wave grid
  v8f vz = {};
  v8f acc[2][4];
#pragma unroll
  for (int i = 0; i < 2; ++i)
#pragma unroll
    for (int j = 0; j < 4; ++j) acc[i][j] = vz;

  int lr = threadIdx.x >> 1;                   // 0..127 tile row
  int lk = (threadIdx.x & 1) * 16;             // 0 / 16

  for (int k0 = 0; k0 < K; k0 += 32) {
    __syncthreads();
    {
      v8h z = {};
      int gm = m0 + lr;
      v8h a0 = z, a1 = z;
      if (gm < M) {
        const _Float16* src = A + (size_t)gm * K + k0 + lk;
        a0 = *(const v8h*)src;
        a1 = *(const v8h*)(src + 8);
      }
      *(v8h*)&sA[lr * LDT + lk] = a0;
      *(v8h*)&sA[lr * LDT + lk + 8] = a1;
      int gn = n0 + lr;
      v8h b0 = z, b1 = z;
      if (gn < N) {
        const _Float16* src = W + (size_t)gn * K + k0 + lk;
        b0 = *(const v8h*)src;
        b1 = *(const v8h*)(src + 8);
      }
      *(v8h*)&sB[lr * LDT + lk] = b0;
      *(v8h*)&sB[lr * LDT + lk + 8] = b1;
    }
    __syncthreads();
    v16h af0 = frag_a(sA, wm * 32, 0, LDT);
    v16h af1 = frag_a(sA, wm * 32 + 16, 0, LDT);
    v16h bf[4];
#pragma unroll
    for (int j = 0; j < 4; ++j) bf[j] = frag_b(sB, wn * 64 + j * 16, 0, LDT);
#pragma unroll
    for (int j = 0; j < 4; ++j) {
      acc[0][j] = wmma16(af0, bf[j], acc[0][j]);
      acc[1][j] = wmma16(af1, bf[j], acc[1][j]);
    }
  }

  int lane = threadIdx.x & 31;
  int rbase = (lane >> 4) * 8;
  int cn = lane & 15;
#pragma unroll
  for (int i = 0; i < 2; ++i) {
#pragma unroll
    for (int j = 0; j < 4; ++j) {
      int n = n0 + wn * 64 + j * 16 + cn;
      if (n >= N) continue;
      float bv = bias ? bias[n] : 0.0f;
#pragma unroll
      for (int r = 0; r < 8; ++r) {
        int m = m0 + wm * 32 + i * 16 + rbase + r;
        if (m >= M) continue;
        float v = acc[i][j][r] + bv;
        size_t off = (size_t)m * N + n;
        if (EP == 0) outH[off] = (_Float16)v;
        else if (EP == 1) outH[off] = (_Float16)gelu_exact(v);
        else if (EP == 2) outF[off] = resid[off] + ls[n] * v;
        else outF[off] = v;
      }
    }
  }
}

// ---------------------------------------------------------------------------
// Flash attention: qkv f16 [B*N, 3072] (Q|K|V, each [H,64]) -> out f16 [B*N, 1024]
// Block = 128 threads (4 waves), 64 query rows; key step 32; online softmax.
__global__ __launch_bounds__(128) void k_attn(const _Float16* __restrict__ qkv,
                                              _Float16* __restrict__ out, float scale) {
  const int QLD = 3 * VD;
  int b = blockIdx.x >> 4, h = blockIdx.x & 15;
  int q0 = blockIdx.y * 64;
  int wave = threadIdx.x >> 5, lane = threadIdx.x & 31;

  __shared__ _Float16 sQ[64 * 72];
  __shared__ _Float16 sK[32 * 72];
  __shared__ _Float16 sV[32 * 72];
  __shared__ _Float16 sP[4][16 * 40];

  const _Float16* base = qkv + (size_t)b * VN * QLD + h * VHD;

  {  // stage Q tile [64 x 64]
    int r = threadIdx.x >> 1;
    int part = (threadIdx.x & 1) * 32;
    bool ok = (q0 + r) < VN;
    v8h z = {};
    const _Float16* src = base + (size_t)(q0 + r) * QLD + part;
#pragma unroll
    for (int i = 0; i < 4; ++i)
      *(v8h*)&sQ[r * 72 + part + i * 8] = ok ? *(const v8h*)(src + i * 8) : z;
  }
  __syncthreads();
  v16h qa0 = frag_a(sQ, wave * 16, 0, 72);
  v16h qa1 = frag_a(sQ, wave * 16, 32, 72);

  v8f vz = {};
  v8f o[4];
#pragma unroll
  for (int t = 0; t < 4; ++t) o[t] = vz;
  float mst[8], lst[8];
#pragma unroll
  for (int r = 0; r < 8; ++r) { mst[r] = -3.0e38f; lst[r] = 0.0f; }

  for (int j0 = 0; j0 < VN; j0 += 32) {
    __syncthreads();
    {  // stage K,V tiles [32 x 64]
      int r = threadIdx.x >> 2;
      int part = (threadIdx.x & 3) * 16;
      bool ok = (j0 + r) < VN;
      v8h z = {};
      const _Float16* ks = base + (size_t)(j0 + r) * QLD + VD + part;
      const _Float16* vs = ks + VD;
      *(v8h*)&sK[r * 72 + part] = ok ? *(const v8h*)ks : z;
      *(v8h*)&sK[r * 72 + part + 8] = ok ? *(const v8h*)(ks + 8) : z;
      *(v8h*)&sV[r * 72 + part] = ok ? *(const v8h*)vs : z;
      *(v8h*)&sV[r * 72 + part + 8] = ok ? *(const v8h*)(vs + 8) : z;
    }
    __syncthreads();

    v8f s0 = vz, s1 = vz;
    {
      v16h kb;
      kb = frag_b(sK, 0, 0, 72);  s0 = wmma16(qa0, kb, s0);
      kb = frag_b(sK, 0, 32, 72); s0 = wmma16(qa1, kb, s0);
      kb = frag_b(sK, 16, 0, 72); s1 = wmma16(qa0, kb, s1);
      kb = frag_b(sK, 16, 32, 72); s1 = wmma16(qa1, kb, s1);
    }
    int key0 = j0 + (lane & 15);
    float alpha[8];
#pragma unroll
    for (int r = 0; r < 8; ++r) {
      float a = s0[r] * scale; if (key0 >= VN) a = -3.0e38f;
      float c = s1[r] * scale; if (key0 + 16 >= VN) c = -3.0e38f;
      float mx = fmaxf(a, c);
      mx = fmaxf(mx, __shfl_xor(mx, 1, 16));
      mx = fmaxf(mx, __shfl_xor(mx, 2, 16));
      mx = fmaxf(mx, __shfl_xor(mx, 4, 16));
      mx = fmaxf(mx, __shfl_xor(mx, 8, 16));
      float mnew = fmaxf(mst[r], mx);
      alpha[r] = expf(mst[r] - mnew);
      mst[r] = mnew;
      float p0 = expf(a - mnew);
      float p1 = expf(c - mnew);
      float rs = p0 + p1;
      rs += __shfl_xor(rs, 1, 16);
      rs += __shfl_xor(rs, 2, 16);
      rs += __shfl_xor(rs, 4, 16);
      rs += __shfl_xor(rs, 8, 16);
      lst[r] = lst[r] * alpha[r] + rs;
      s0[r] = p0; s1[r] = p1;
    }
#pragma unroll
    for (int t = 0; t < 4; ++t)
#pragma unroll
      for (int r = 0; r < 8; ++r) o[t][r] *= alpha[r];

    // stage P (16x32) through LDS to get A-fragment layout
    _Float16* P = sP[wave];
    int prow = (lane >> 4) * 8;
#pragma unroll
    for (int r = 0; r < 8; ++r) {
      P[(prow + r) * 40 + (lane & 15)] = (_Float16)s0[r];
      P[(prow + r) * 40 + 16 + (lane & 15)] = (_Float16)s1[r];
    }
    __syncthreads();
    v16h pa = frag_a(P, 0, 0, 40);
#pragma unroll
    for (int t = 0; t < 4; ++t) {
      v16h vb = frag_b_kmajor(sV, t * 16, 72);
      o[t] = wmma16(pa, vb, o[t]);
    }
  }

  int prow = (lane >> 4) * 8, cn = lane & 15;
#pragma unroll
  for (int t = 0; t < 4; ++t) {
#pragma unroll
    for (int r = 0; r < 8; ++r) {
      int m = q0 + wave * 16 + prow + r;
      if (m < VN) {
        float val = o[t][r] / lst[r];
        out[(size_t)(b * VN + m) * VD + h * VHD + t * 16 + cn] = (_Float16)val;
      }
    }
  }
}

// ---------------------------------------------------------------------------
template <int EP>
static void launch_gemm(const _Float16* A, const _Float16* W, const float* bias,
                        const float* ls, const float* resid, float* outF,
                        _Float16* outH, int M, int N, int K, hipStream_t stream) {
  dim3 g((N + TN - 1) / TN, (M + TM - 1) / TM);
  k_gemm<EP><<<g, 256, 0, stream>>>(A, W, bias, ls, resid, outF, outH, M, N, K);
}

extern "C" void kernel_launch(void* const* d_in, const int* in_sizes, int n_in,
                              void* d_out, int out_size, void* d_ws, size_t ws_size,
                              hipStream_t stream) {
  (void)in_sizes; (void)n_in; (void)out_size; (void)ws_size;
  const float* pix    = (const float*)d_in[0];
  const float* cls    = (const float*)d_in[1];
  const float* patchw = (const float*)d_in[2];
  const float* patchb = (const float*)d_in[3];
  const float* pos    = (const float*)d_in[4];
  const float* n1w    = (const float*)d_in[5];
  const float* n1b    = (const float*)d_in[6];
  const float* qkvw   = (const float*)d_in[7];
  const float* qkvb   = (const float*)d_in[8];
  const float* projw  = (const float*)d_in[9];
  const float* projb  = (const float*)d_in[10];
  const float* ls1    = (const float*)d_in[11];
  const float* n2w    = (const float*)d_in[12];
  const float* n2b    = (const float*)d_in[13];
  const float* f1w    = (const float*)d_in[14];
  const float* f1b    = (const float*)d_in[15];
  const float* f2w    = (const float*)d_in[16];
  const float* f2b    = (const float*)d_in[17];
  const float* ls2    = (const float*)d_in[18];
  const float* lnw    = (const float*)d_in[19];
  const float* lnb    = (const float*)d_in[20];
  const float* m1w    = (const float*)d_in[21];
  const float* m1b    = (const float*)d_in[22];
  const float* m2w    = (const float*)d_in[23];
  const float* m2b    = (const float*)d_in[24];

  // workspace carving
  char* ws = (char*)d_ws;
  size_t off = 0;
  auto carve = [&](size_t bytes) -> char* {
    off = (off + 255) & ~size_t(255);
    char* p = ws + off;
    off += bytes;
    return p;
  };
  _Float16* wPatch = (_Float16*)carve((size_t)VD * KPAT * 2);
  _Float16* wQkv   = (_Float16*)carve((size_t)4 * 3 * VD * VD * 2);
  _Float16* wProj  = (_Float16*)carve((size_t)4 * VD * VD * 2);
  _Float16* wFc1   = (_Float16*)carve((size_t)4 * VI * VD * 2);
  _Float16* wFc2   = (_Float16*)carve((size_t)4 * VD * VI * 2);
  _Float16* wM1    = (_Float16*)carve((size_t)LLMD * 4096 * 2);
  _Float16* wM2    = (_Float16*)carve((size_t)LLMD * LLMD * 2);
  _Float16* im2c   = (_Float16*)carve((size_t)VB * 1024 * KPAT * 2);
  float*    patcho = (float*)carve((size_t)VB * 1024 * VD * 4);
  float*    x      = (float*)carve((size_t)VM * VD * 4);
  _Float16* h16    = (_Float16*)carve((size_t)VM * VD * 2);
  _Float16* qkv16  = (_Float16*)carve((size_t)VM * 3 * VD * 2);
  _Float16* att16  = (_Float16*)carve((size_t)VM * VD * 2);
  _Float16* fc1o   = (_Float16*)carve((size_t)VM * VI * 2);
  _Float16* shuf16 = (_Float16*)carve((size_t)1024 * 4096 * 2);
  _Float16* proj1  = (_Float16*)carve((size_t)1024 * LLMD * 2);

  auto cvt = [&](const float* s, _Float16* d, size_t n) {
    k_cvt<<<(unsigned)((n + 1023) / 1024), 256, 0, stream>>>(s, d, (int)n);
  };
  cvt(qkvw, wQkv, (size_t)4 * 3 * VD * VD);
  cvt(projw, wProj, (size_t)4 * VD * VD);
  cvt(f1w, wFc1, (size_t)4 * VI * VD);
  cvt(f2w, wFc2, (size_t)4 * VD * VI);
  cvt(m1w, wM1, (size_t)LLMD * 4096);
  cvt(m2w, wM2, (size_t)LLMD * LLMD);
  k_cvt_patchw<<<(VD * KPAT + 255) / 256, 256, 0, stream>>>(patchw, wPatch);

  // patch embedding: im2col + GEMM (f32 out + bias)
  k_im2col<<<(VB * 1024 * KPAT + 255) / 256, 256, 0, stream>>>(pix, im2c);
  launch_gemm<3>(im2c, wPatch, patchb, nullptr, nullptr, patcho, nullptr,
                 VB * 1024, VD, KPAT, stream);
  k_build_x<<<(VM * VD + 255) / 256, 256, 0, stream>>>(cls, pos, patcho, x);

  // transformer layers
  for (int l = 0; l < 4; ++l) {
    k_layernorm<<<VM, 256, 0, stream>>>(x, n1w + l * VD, n1b + l * VD, h16, VD);
    launch_gemm<0>(h16, wQkv + (size_t)l * 3 * VD * VD, qkvb + l * 3 * VD,
                   nullptr, nullptr, nullptr, qkv16, VM, 3 * VD, VD, stream);
    k_attn<<<dim3(VB * VH, (VN + 63) / 64), 128, 0, stream>>>(qkv16, att16, 0.125f);
    launch_gemm<2>(att16, wProj + (size_t)l * VD * VD, projb + l * VD,
                   ls1 + l * VD, x, x, nullptr, VM, VD, VD, stream);
    k_layernorm<<<VM, 256, 0, stream>>>(x, n2w + l * VD, n2b + l * VD, h16, VD);
    launch_gemm<1>(h16, wFc1 + (size_t)l * VI * VD, f1b + l * VI,
                   nullptr, nullptr, nullptr, fc1o, VM, VI, VD, stream);
    launch_gemm<2>(fc1o, wFc2 + (size_t)l * VD * VI, f2b + l * VD,
                   ls2 + l * VD, x, x, nullptr, VM, VD, VI, stream);
  }

  // pixel shuffle + LN + projector MLP
  k_shuffle_ln<<<VB * 256, 256, 0, stream>>>(x, lnw, lnb, shuf16);
  launch_gemm<1>(shuf16, wM1, m1b, nullptr, nullptr, nullptr, proj1,
                 1024, LLMD, 4096, stream);
  launch_gemm<3>(proj1, wM2, m2b, nullptr, nullptr, (float*)d_out, nullptr,
                 1024, LLMD, LLMD, stream);
}
